// NTM_53137335386347
// MI455X (gfx1250) — matrix operand order repaired and checked by reading, not compiled, not run on hardware
//
#include <hip/hip_runtime.h>
#include <hip/hip_bf16.h>

typedef __attribute__((ext_vector_type(16))) _Float16 v16h;
typedef __attribute__((ext_vector_type(8)))  _Float16 v8h;
typedef __attribute__((ext_vector_type(8)))  float    v8f;

#define BB   64
#define TT   64
#define INF  64
#define CC   256
#define NN   1024
#define MV   64
#define OUTF 64
#define HWP  208   // 3*MV+6=198 padded to 16
#define HRP  80    // MV+6=70 padded to 16
#define EPSF 1e-8f

// ---------------- small math helpers ----------------
__device__ __forceinline__ float sigmoidf_(float x) { return 1.f / (1.f + __expf(-x)); }
__device__ __forceinline__ float softplusf_(float x) {
  return fmaxf(x, 0.f) + log1pf(__expf(-fabsf(x)));
}

// block-wide reduction for 256-thread (8-wave, wave32) blocks. OP: 0=max, 1=sum
template <int OP>
__device__ __forceinline__ float block_reduce(float v, float* red) {
  int lane = threadIdx.x & 31;
  int wid  = threadIdx.x >> 5;
#pragma unroll
  for (int o = 16; o > 0; o >>= 1) {
    float other = __shfl_xor(v, o);
    v = OP ? (v + other) : fmaxf(v, other);
  }
  if (lane == 0) red[wid] = v;
  __syncthreads();
  if (wid == 0) {
    float t = (lane < 8) ? red[lane] : (OP ? 0.f : -3.402823466e38f);
#pragma unroll
    for (int o = 4; o > 0; o >>= 1) {
      float other = __shfl_xor(t, o);
      t = OP ? (t + other) : fmaxf(t, other);
    }
    if (lane == 0) red[0] = t;
  }
  __syncthreads();
  float res = red[0];
  __syncthreads();
  return res;
}

// Pack one B-fragment element index: packed[((ct*KS + ks)*32 + lane)*16 + e]
//   holds W[(ks*32 + (lane>>4)*8 + (e&7) + (e>>3)*16) * ncol + ct*16 + (lane&15)]  (zero-padded cols)
__device__ __forceinline__ void pack_b(const float* __restrict__ W, _Float16* __restrict__ P,
                                       int KS, int CT, int ncol, int idx, int stride) {
  int total = CT * KS * 512;
  for (int i = idx; i < total; i += stride) {
    int e    = i & 15;
    int lane = (i >> 4) & 31;
    int rest = i >> 9;            // ct*KS + ks
    int ks = rest % KS, ct = rest / KS;
    int kk  = ks * 32 + ((lane >> 4) << 3) + (e & 7) + ((e >> 3) << 4);
    int col = ct * 16 + (lane & 15);
    P[i] = (_Float16)((col < ncol) ? W[kk * ncol + col] : 0.f);
  }
}

// Load A fragment (16x32 f16, ISA layout) from row-major f16, row base already
// offset to (m, k0): elements 0..7 at +kh, 8..15 at +16+kh  (kh = (lane>>4)*8)
__device__ __forceinline__ v16h load_a(const _Float16* __restrict__ rowk) {
  v8h lo = *(const v8h*)(rowk);
  v8h hi = *(const v8h*)(rowk + 16);
  v16h a;
#pragma unroll
  for (int e = 0; e < 8; ++e) { a[e] = lo[e]; a[e + 8] = hi[e]; }
  return a;
}

// ---------------- init: state biases, x->f16, pack weight fragments ----------------
__global__ __launch_bounds__(256) void k_init(
    const float* __restrict__ x, const float* __restrict__ r_bias,
    const float* __restrict__ w_bias, const float* __restrict__ M_bias,
    _Float16* __restrict__ x16, _Float16* __restrict__ r16,
    float* __restrict__ w, float* __restrict__ M,
    const float* __restrict__ Wc, const float* __restrict__ Wr,
    const float* __restrict__ Ww, const float* __restrict__ Wf,
    _Float16* __restrict__ PWc, _Float16* __restrict__ PWr,
    _Float16* __restrict__ PWw, _Float16* __restrict__ PWf) {
  int idx = blockIdx.x * blockDim.x + threadIdx.x;
  int stride = gridDim.x * blockDim.x;
  for (int i = idx; i < BB * NN * MV; i += stride) M[i] = M_bias[i & (NN * MV - 1)];
  for (int i = idx; i < BB * MV; i += stride) r16[i] = (_Float16)r_bias[i & (MV - 1)];
  for (int i = idx; i < BB * NN; i += stride) w[i] = w_bias[i & (NN - 1)];
  for (int i = idx; i < BB * TT * INF; i += stride) x16[i] = (_Float16)x[i];
  pack_b(Wc, PWc, /*KS=*/4,  /*CT=*/16, /*ncol=*/CC,         idx, stride);
  pack_b(Ww, PWw, /*KS=*/8,  /*CT=*/13, /*ncol=*/3 * MV + 6, idx, stride);
  pack_b(Wr, PWr, /*KS=*/8,  /*CT=*/5,  /*ncol=*/MV + 6,     idx, stride);
  pack_b(Wf, PWf, /*KS=*/10, /*CT=*/4,  /*ncol=*/OUTF,       idx, stride);
}

// ---------------- controller: c = tanh([x_t, r] @ Wc + bc) ----------------
// [64,128]@[128,256] -> 4x16 = 64 wave tiles; grid 8 x 256 threads
__global__ __launch_bounds__(256) void k_ctrl(
    const _Float16* __restrict__ x16, const _Float16* __restrict__ r16,
    const _Float16* __restrict__ PWc, const float* __restrict__ bc,
    _Float16* __restrict__ c16, int t) {
  int wgid = blockIdx.x * 8 + (threadIdx.x >> 5);  // 0..63
  int lane = threadIdx.x & 31;
  int tile_m = (wgid >> 4) << 4;
  int ct     = wgid & 15;
  int m  = tile_m + (lane & 15);
  int kh = (lane >> 4) << 3;
  const v16h* Bf = (const v16h*)PWc;
  v8f acc = {};
#pragma unroll
  for (int ks = 0; ks < 4; ++ks) {
    int k0 = ks * 32;
    const _Float16* abase = (k0 < INF)
        ? (x16 + m * (TT * INF) + t * INF + k0 + kh)
        : (r16 + m * MV + (k0 - INF) + kh);
    v16h a = load_a(abase);
    v16h b = Bf[(ct * 4 + ks) * 32 + lane];
    acc = __builtin_amdgcn_wmma_f32_16x16x32_f16(false, a, false, b, (short)0, acc,
                                                 false, false);
  }
  int col   = ct * 16 + (lane & 15);
  int rbase = tile_m + ((lane >> 4) << 3);
  float bias = bc[col];
#pragma unroll
  for (int i = 0; i < 8; ++i)
    c16[(rbase + i) * CC + col] = (_Float16)tanhf(acc[i] + bias);
}

// ---------------- heads: hw = c@Ww + bw (208 cols), hr = c@Wr + br (80 cols) ----------------
// 4 row tiles x (13 + 5) col tiles = 72 wave tiles; grid 9 x 256
__global__ __launch_bounds__(256) void k_heads(
    const _Float16* __restrict__ c16, const _Float16* __restrict__ PWw,
    const _Float16* __restrict__ PWr, const float* __restrict__ bw,
    const float* __restrict__ br, float* __restrict__ hw, float* __restrict__ hr) {
  int wgid = blockIdx.x * 8 + (threadIdx.x >> 5);  // 0..71
  int lane = threadIdx.x & 31;
  int rt = wgid / 18;
  int ct = wgid - rt * 18;
  bool is_w = (ct < 13);
  int ctl = is_w ? ct : (ct - 13);
  const v16h* Bf = (const v16h*)(is_w ? PWw : PWr);
  int m  = (rt << 4) + (lane & 15);
  int kh = (lane >> 4) << 3;
  v8f acc = {};
#pragma unroll
  for (int ks = 0; ks < 8; ++ks) {
    v16h a = load_a(c16 + m * CC + ks * 32 + kh);
    v16h b = Bf[(ctl * 8 + ks) * 32 + lane];
    acc = __builtin_amdgcn_wmma_f32_16x16x32_f16(false, a, false, b, (short)0, acc,
                                                 false, false);
  }
  int col   = ctl * 16 + (lane & 15);
  int rbase = (rt << 4) + ((lane >> 4) << 3);
  float bias;
  float* dst;
  int ldd;
  if (is_w) { bias = (col < 3 * MV + 6) ? bw[col] : 0.f; dst = hw; ldd = HWP; }
  else      { bias = (col < MV + 6) ? br[col] : 0.f;     dst = hr; ldd = HRP; }
#pragma unroll
  for (int i = 0; i < 8; ++i) dst[(rbase + i) * ldd + col] = acc[i] + bias;
}

// ---------------- NTM addressing (shared by write & read heads) ----------------
__device__ void ntm_address(const float* __restrict__ h,
                            const float* __restrict__ wprev,
                            const float* __restrict__ Mb,
                            float* wg, float* wout, float* red, float* sk,
                            float* sp) {
  int tid = threadIdx.x;
  if (tid < MV) sk[tid] = tanhf(h[tid]);
  if (tid == 0) {
    sp[0] = softplusf_(h[MV]);       // beta
    sp[1] = sigmoidf_(h[MV + 1]);    // g
    float s0 = h[MV + 2], s1 = h[MV + 3], s2 = h[MV + 4];
    float mx = fmaxf(s0, fmaxf(s1, s2));
    float e0 = __expf(s0 - mx), e1 = __expf(s1 - mx), e2 = __expf(s2 - mx);
    float inv = 1.f / (e0 + e1 + e2);
    sp[2] = e0 * inv; sp[3] = e1 * inv; sp[4] = e2 * inv;
    sp[5] = 1.f + softplusf_(h[MV + 5]);  // gamma
  }
  __syncthreads();
  float p = (tid < MV) ? sk[tid] * sk[tid] : 0.f;
  float knrm = sqrtf(block_reduce<1>(p, red));
  float beta = sp[0], g = sp[1], gamma = sp[5];
  float s0 = sp[2], s1 = sp[3], s2 = sp[4];

  float sim[4];
#pragma unroll
  for (int i = 0; i < 4; ++i) {
    int n = tid + i * 256;
    const float4* Mr = (const float4*)(Mb + (size_t)n * MV);
    __builtin_prefetch(Mb + (size_t)(n + 1) * MV, 0, 1);
    float dot = 0.f, nrm = 0.f;
#pragma unroll
    for (int j = 0; j < MV / 4; ++j) {
      float4 mv = Mr[j];
      float4 kv = ((const float4*)sk)[j];
      dot += mv.x * kv.x + mv.y * kv.y + mv.z * kv.z + mv.w * kv.w;
      nrm += mv.x * mv.x + mv.y * mv.y + mv.z * mv.z + mv.w * mv.w;
    }
    sim[i] = beta * dot / (sqrtf(nrm) * knrm + EPSF);
  }
  float lmax = fmaxf(fmaxf(sim[0], sim[1]), fmaxf(sim[2], sim[3]));
  float gmax = block_reduce<0>(lmax, red);
  float ex[4], lsum = 0.f;
#pragma unroll
  for (int i = 0; i < 4; ++i) { ex[i] = __expf(sim[i] - gmax); lsum += ex[i]; }
  float inv = 1.f / block_reduce<1>(lsum, red);
#pragma unroll
  for (int i = 0; i < 4; ++i) {
    int n = tid + i * 256;
    float wc = ex[i] * inv;
    wg[n] = g * wc + (1.f - g) * wprev[n];
  }
  __syncthreads();
  float wp[4], lsum2 = 0.f;
#pragma unroll
  for (int i = 0; i < 4; ++i) {
    int n = tid + i * 256;
    float wsv = s0 * wg[(n + 1) & (NN - 1)] + s1 * wg[n] + s2 * wg[(n + NN - 1) & (NN - 1)];
    wp[i] = __expf(gamma * __logf(wsv + EPSF));
    lsum2 += wp[i];
  }
  float inv2 = 1.f / block_reduce<1>(lsum2, red);
#pragma unroll
  for (int i = 0; i < 4; ++i) wout[tid + i * 256] = wp[i] * inv2;
  __syncthreads();
}

// ---------------- write addressing + memory erase/add update ----------------
__global__ __launch_bounds__(256) void k_wraddr(
    const float* __restrict__ hw, const float* __restrict__ wprev,
    float* __restrict__ M, float* __restrict__ ww) {
  __shared__ float wg[NN], wout[NN], sk[MV], se[MV], sa[MV], red[32], sp[8];
  int b = blockIdx.x, tid = threadIdx.x;
  const float* h = hw + b * HWP;
  if (tid < MV) {
    se[tid] = sigmoidf_(h[MV + 6 + tid]);
    sa[tid] = tanhf(h[2 * MV + 6 + tid]);
  }
  float* Mb = M + (size_t)b * NN * MV;
  ntm_address(h, wprev + b * NN, Mb, wg, wout, red, sk, sp);
#pragma unroll
  for (int i = 0; i < 4; ++i) ww[b * NN + tid + i * 256] = wout[tid + i * 256];
  for (int idx = tid; idx < NN * (MV / 4); idx += 256) {
    int n = idx >> 4, q = idx & 15;
    float wv = wout[n];
    float4* pm = (float4*)(Mb + (size_t)n * MV + q * 4);
    float4 mv = *pm;
    float4 ev = ((const float4*)se)[q];
    float4 av = ((const float4*)sa)[q];
    mv.x = mv.x * (1.f - wv * ev.x) + wv * av.x;
    mv.y = mv.y * (1.f - wv * ev.y) + wv * av.y;
    mv.z = mv.z * (1.f - wv * ev.z) + wv * av.z;
    mv.w = mv.w * (1.f - wv * ev.w) + wv * av.w;
    *pm = mv;
  }
}

// ---------------- read addressing + read vector (r emitted as f16) ----------------
__global__ __launch_bounds__(256) void k_rdaddr(
    const float* __restrict__ hr, const float* __restrict__ wprev,
    const float* __restrict__ M, float* __restrict__ w, _Float16* __restrict__ r16) {
  __shared__ float wg[NN], wout[NN], sk[MV], red[32], sp[8];
  __shared__ float racc[256];
  int b = blockIdx.x, tid = threadIdx.x;
  const float* Mb = M + (size_t)b * NN * MV;
  ntm_address(hr + b * HRP, wprev + b * NN, Mb, wg, wout, red, sk, sp);
#pragma unroll
  for (int i = 0; i < 4; ++i) w[b * NN + tid + i * 256] = wout[tid + i * 256];
  int m = tid & (MV - 1), noff = tid >> 6;
  float acc = 0.f;
  for (int n = noff; n < NN; n += 4) {
    __builtin_prefetch(Mb + (size_t)(n + 4) * MV + m, 0, 1);
    acc += wout[n] * Mb[(size_t)n * MV + m];
  }
  racc[tid] = acc;
  __syncthreads();
  if (tid < MV)
    r16[b * MV + tid] =
        (_Float16)(racc[tid] + racc[tid + 64] + racc[tid + 128] + racc[tid + 192]);
}

// ---------------- output: out_t = sigmoid([c, r] @ Wf + bf) ----------------
// [64,320]@[320,64] -> 4x4 = 16 wave tiles; grid 2 x 256
__global__ __launch_bounds__(256) void k_out(
    const _Float16* __restrict__ c16, const _Float16* __restrict__ r16,
    const _Float16* __restrict__ PWf, const float* __restrict__ bf,
    float* __restrict__ out, int t) {
  int wgid = blockIdx.x * 8 + (threadIdx.x >> 5);  // 0..15
  int lane = threadIdx.x & 31;
  int tile_m = (wgid >> 2) << 4;
  int ct     = wgid & 3;
  int m  = tile_m + (lane & 15);
  int kh = (lane >> 4) << 3;
  const v16h* Bf = (const v16h*)PWf;
  v8f acc = {};
#pragma unroll
  for (int ks = 0; ks < 10; ++ks) {
    int k0 = ks * 32;
    const _Float16* abase = (k0 < CC)
        ? (c16 + m * CC + k0 + kh)
        : (r16 + m * MV + (k0 - CC) + kh);
    v16h a = load_a(abase);
    v16h b = Bf[(ct * 10 + ks) * 32 + lane];
    acc = __builtin_amdgcn_wmma_f32_16x16x32_f16(false, a, false, b, (short)0, acc,
                                                 false, false);
  }
  int col   = ct * 16 + (lane & 15);
  int rbase = tile_m + ((lane >> 4) << 3);
  float bias = bf[col];
#pragma unroll
  for (int i = 0; i < 8; ++i)
    out[(rbase + i) * (TT * OUTF) + t * OUTF + col] = sigmoidf_(acc[i] + bias);
}

// ---------------- host launcher ----------------
extern "C" void kernel_launch(void* const* d_in, const int* in_sizes, int n_in,
                              void* d_out, int out_size, void* d_ws, size_t ws_size,
                              hipStream_t stream) {
  const float* x      = (const float*)d_in[0];
  const float* Wc     = (const float*)d_in[1];
  const float* bc     = (const float*)d_in[2];
  const float* Wr     = (const float*)d_in[3];
  const float* br     = (const float*)d_in[4];
  const float* Ww     = (const float*)d_in[5];
  const float* bw     = (const float*)d_in[6];
  const float* Wf     = (const float*)d_in[7];
  const float* bf     = (const float*)d_in[8];
  const float* r_bias = (const float*)d_in[9];
  const float* w_bias = (const float*)d_in[10];
  const float* M_bias = (const float*)d_in[11];
  float* out = (float*)d_out;

  char* ws = (char*)d_ws;
  size_t off = 0;
  float* M  = (float*)(ws + off); off += (size_t)BB * NN * MV * 4;   // 16 MB
  float* w  = (float*)(ws + off); off += (size_t)BB * NN * 4;
  float* ww = (float*)(ws + off); off += (size_t)BB * NN * 4;
  float* hw = (float*)(ws + off); off += (size_t)BB * HWP * 4;
  float* hr = (float*)(ws + off); off += (size_t)BB * HRP * 4;
  _Float16* x16 = (_Float16*)(ws + off); off += (size_t)BB * TT * INF * 2;
  _Float16* r16 = (_Float16*)(ws + off); off += (size_t)BB * MV * 2;
  _Float16* c16 = (_Float16*)(ws + off); off += (size_t)BB * CC * 2;
  _Float16* PWc = (_Float16*)(ws + off); off += (size_t)16 * 4 * 512 * 2;
  _Float16* PWw = (_Float16*)(ws + off); off += (size_t)13 * 8 * 512 * 2;
  _Float16* PWr = (_Float16*)(ws + off); off += (size_t)5 * 8 * 512 * 2;
  _Float16* PWf = (_Float16*)(ws + off); off += (size_t)4 * 10 * 512 * 2;

  k_init<<<2048, 256, 0, stream>>>(x, r_bias, w_bias, M_bias, x16, r16, w, M,
                                   Wc, Wr, Ww, Wf, PWc, PWr, PWw, PWf);
  for (int t = 0; t < TT; ++t) {
    k_ctrl<<<8, 256, 0, stream>>>(x16, r16, PWc, bc, c16, t);
    k_heads<<<9, 256, 0, stream>>>(c16, PWw, PWr, bw, br, hw, hr);
    k_wraddr<<<BB, 256, 0, stream>>>(hw, w, M, ww);
    k_rdaddr<<<BB, 256, 0, stream>>>(hr, ww, M, w, r16);
    k_out<<<2, 256, 0, stream>>>(c16, r16, PWf, bf, out, t);
  }
}
